// DiffMultiHeadedCrossAttention_51522427683066
// MI455X (gfx1250) — matrix operand order repaired and checked by reading, not compiled, not run on hardware
//
#include <hip/hip_runtime.h>

#define HEADS 16
#define HID   1024
#define DH    64
#define HALFD 32
#define BB    2
#define TT    2048
#define NN    2048

typedef __attribute__((ext_vector_type(16))) __bf16 v16bf;
typedef __attribute__((ext_vector_type(8)))  float  v8f;
typedef unsigned short u16;
typedef __attribute__((ext_vector_type(8))) u16 u16x8;
typedef __attribute__((ext_vector_type(4))) u16 u16x4;

union Frag {
    u16x8 h[2];
    u16   s[16];
    v16bf bf;
};

static __device__ __forceinline__ u16 f2bf(float f) {
    union { __bf16 b; u16 u; } cv;
    cv.b = (__bf16)f;       // native conversion (hardware cvt where available)
    return cv.u;
}

// ---------------------------------------------------------------------------
// Kernel 0: bulk f32 -> bf16 conversion (memory-bound, one pass).
// ---------------------------------------------------------------------------
__global__ __launch_bounds__(256) void cvt_bf16_kernel(
    const float* __restrict__ in, u16* __restrict__ out, int n4)
{
    const int i = blockIdx.x * 256 + threadIdx.x;
    if (i >= n4) return;
    const float4 f = ((const float4*)in)[i];
    u16x4 o;
    o.x = f2bf(f.x); o.y = f2bf(f.y); o.z = f2bf(f.z); o.w = f2bf(f.w);
    ((u16x4*)out)[i] = o;
}

// ---------------------------------------------------------------------------
// Kernel 1: pure-bf16 projection GEMM  Y[m,n] = sum_k X[m,k] * W[n,k].
// rows = 4096 (B*T), K = N = HID. Wave computes a 32x64 tile
// (2 M-tiles x 4 N-tiles -> 8 WMMAs per 32-wide K chunk).
// ---------------------------------------------------------------------------
__global__ __launch_bounds__(256) void proj_gemm_bf16(
    const u16* __restrict__ X, const u16* __restrict__ W,
    u16* __restrict__ Y)
{
    const int wid  = (blockIdx.x * 256 + threadIdx.x) >> 5;
    const int lane = threadIdx.x & 31;
    const int l16  = lane & 15;
    const int half = lane >> 4;
    const int mt   = wid >> 4;   // 0..127 (32-row tiles)
    const int ng   = wid & 15;   // 0..15  (64-col groups)

    const u16* xrow0 = X + (size_t)(mt * 32 + l16) * HID;
    const u16* xrow1 = xrow0 + (size_t)16 * HID;

    v8f acc[2][4] = {{v8f{}, v8f{}, v8f{}, v8f{}},
                     {v8f{}, v8f{}, v8f{}, v8f{}}};

    for (int kc = 0; kc < HID; kc += 32) {
        Frag a0, a1;
        const int ka = kc + half * 8;
        a0.h[0] = *(const u16x8*)(xrow0 + ka);
        a0.h[1] = *(const u16x8*)(xrow0 + ka + 16);
        a1.h[0] = *(const u16x8*)(xrow1 + ka);
        a1.h[1] = *(const u16x8*)(xrow1 + ka + 16);

        #pragma unroll
        for (int j = 0; j < 4; ++j) {
            const u16* wrow = W + (size_t)(ng * 64 + j * 16 + l16) * HID
                                + kc + half * 8;
            Frag bfr;
            bfr.h[0] = *(const u16x8*)(wrow);
            bfr.h[1] = *(const u16x8*)(wrow + 16);
            acc[0][j] = __builtin_amdgcn_wmma_f32_16x16x32_bf16(
                false, a0.bf, false, bfr.bf, (short)0, acc[0][j], false, false);
            acc[1][j] = __builtin_amdgcn_wmma_f32_16x16x32_bf16(
                false, a1.bf, false, bfr.bf, (short)0, acc[1][j], false, false);
        }
    }

    #pragma unroll
    for (int mi = 0; mi < 2; ++mi) {
        #pragma unroll
        for (int j = 0; j < 4; ++j) {
            const int ncol = ng * 64 + j * 16 + l16;
            #pragma unroll
            for (int r = 0; r < 8; ++r) {
                const int row = mt * 32 + mi * 16 + half * 8 + r;
                Y[(size_t)row * HID + ncol] = f2bf(acc[mi][j][r]);
            }
        }
    }
}

// ---------------------------------------------------------------------------
// CDNA5 async copy: global -> LDS, tracked by ASYNCcnt.
// ---------------------------------------------------------------------------
static __device__ __forceinline__ void async_load_b128(unsigned lds_off,
                                                       const u16* gaddr)
{
    asm volatile("global_load_async_to_lds_b128 %0, %1, off"
                 :: "v"(lds_off), "v"(gaddr) : "memory");
}
static __device__ __forceinline__ void wait_async0()
{
    asm volatile("s_wait_asynccnt 0" ::: "memory");
}

// ---------------------------------------------------------------------------
// Kernel 2: differential flash attention.
// Block = 8 waves, all same (b, head); wave owns a 16-row T tile.
// ---------------------------------------------------------------------------
static __device__ __forceinline__ void softmax_update(
    const v8f& t0, const v8f& t1, float* m, float* l, float* w,
    u16* prow, int half, int l16, float scale)
{
    #pragma unroll
    for (int r = 0; r < 8; ++r) {
        float a  = t0[r] * scale;
        float bq = t1[r] * scale;
        float v  = fmaxf(a, bq);
        #pragma unroll
        for (int off = 1; off < 16; off <<= 1)
            v = fmaxf(v, __shfl_xor(v, off, 32));
        const float mn = fmaxf(m[r], v);
        const float p0 = __expf(a  - mn);
        const float p1 = __expf(bq - mn);
        float ps = p0 + p1;
        #pragma unroll
        for (int off = 1; off < 16; off <<= 1)
            ps += __shfl_xor(ps, off, 32);
        const float wr = __expf(m[r] - mn);
        l[r] = l[r] * wr + ps;
        m[r] = mn;
        w[r] = wr;
        prow[(half * 8 + r) * 32 + l16]      = f2bf(p0);
        prow[(half * 8 + r) * 32 + 16 + l16] = f2bf(p1);
    }
}

__global__ __launch_bounds__(256) void diffattn_kernel(
    const u16* __restrict__ Q, const u16* __restrict__ K,
    const u16* __restrict__ V, float* __restrict__ O,
    const float* __restrict__ lq1, const float* __restrict__ lq2,
    const float* __restrict__ lk1, const float* __restrict__ lk2)
{
    __shared__ u16 ktile[32][64];    // [enc_row][d 0..63] (k1 | k2)
    __shared__ u16 vtileT[64][32];   // transposed: [d][enc_row]
    __shared__ u16 pbuf[8][16][32];  // per-wave p re-layout buffer

    const int tid  = threadIdx.x;
    const int wv   = tid >> 5;
    const int lane = tid & 31;
    const int half = lane >> 4;
    const int l16  = lane & 15;
    const int b0   = half * 8;

    const int tg    = blockIdx.x & 15;
    const int h     = (blockIdx.x >> 4) & 15;
    const int b     = blockIdx.x >> 8;
    const int ttile = tg * 8 + wv;
    const int trow  = ttile * 16 + l16;

    // lambda scalar
    float s1 = 0.f, s2 = 0.f;
    for (int i = 0; i < HALFD; ++i) { s1 += lq1[i] * lk1[i]; s2 += lq2[i] * lk2[i]; }
    const float lam   = __expf(s1) - __expf(s2) + 0.8f;
    const float scale = 0.17677669529663689f;   // 1/sqrt(32)

    // Q fragments (resident for whole loop)
    const u16* qrow = Q + ((size_t)(b * TT) + trow) * HID + h * DH;
    Frag qa1, qa2;
    qa1.h[0] = *(const u16x8*)(qrow + b0);
    qa1.h[1] = *(const u16x8*)(qrow + b0 + 16);
    qa2.h[0] = *(const u16x8*)(qrow + 32 + b0);
    qa2.h[1] = *(const u16x8*)(qrow + 32 + b0 + 16);

    float m1[8], l1[8], w1[8], m2[8], l2[8], w2[8];
    #pragma unroll
    for (int r = 0; r < 8; ++r) {
        m1[r] = -3.0e38f; l1[r] = 0.f;
        m2[r] = -3.0e38f; l2[r] = 0.f;
    }
    v8f acc1[4] = {v8f{}, v8f{}, v8f{}, v8f{}};
    v8f acc2[4] = {v8f{}, v8f{}, v8f{}, v8f{}};

    // staging geometry: thread -> (row 0..31, 8-col group)
    const int srow = tid >> 3;
    const int scg  = (tid & 7) * 8;
    const u16* kbase = K + ((size_t)(b * NN) + srow) * HID + h * DH + scg;
    const u16* vbase = V + ((size_t)(b * NN) + srow) * HID + h * DH + scg;
    const unsigned klds = (unsigned)(size_t)(&ktile[srow][scg]);

    for (int nb = 0; nb < NN; nb += 32) {
        __syncthreads();
        {   // K: async global->LDS copy (ASYNCcnt); V: load + transpose store
            const u16* ksrc = kbase + (size_t)nb * HID;
            const u16* vsrc = vbase + (size_t)nb * HID;
            async_load_b128(klds, ksrc);
            u16x8 vv = *(const u16x8*)vsrc;
            #pragma unroll
            for (int i = 0; i < 8; ++i) vtileT[scg + i][srow] = vv[i];
            if (nb + 32 < NN) {
                __builtin_prefetch(ksrc + (size_t)32 * HID);
                __builtin_prefetch(vsrc + (size_t)32 * HID);
            }
            wait_async0();
        }
        __syncthreads();

        // ---- scores: s1 = q1*k1^T, s2 = q2*k2^T (16x32 each, 4 WMMAs) ----
        Frag kf;
        kf.h[0] = *(const u16x8*)&ktile[l16][half * 8];
        kf.h[1] = *(const u16x8*)&ktile[l16][half * 8 + 16];
        v8f s1t0 = __builtin_amdgcn_wmma_f32_16x16x32_bf16(
            false, qa1.bf, false, kf.bf, (short)0, v8f{}, false, false);

        kf.h[0] = *(const u16x8*)&ktile[16 + l16][half * 8];
        kf.h[1] = *(const u16x8*)&ktile[16 + l16][half * 8 + 16];
        v8f s1t1 = __builtin_amdgcn_wmma_f32_16x16x32_bf16(
            false, qa1.bf, false, kf.bf, (short)0, v8f{}, false, false);

        kf.h[0] = *(const u16x8*)&ktile[l16][32 + half * 8];
        kf.h[1] = *(const u16x8*)&ktile[l16][32 + half * 8 + 16];
        v8f s2t0 = __builtin_amdgcn_wmma_f32_16x16x32_bf16(
            false, qa2.bf, false, kf.bf, (short)0, v8f{}, false, false);

        kf.h[0] = *(const u16x8*)&ktile[16 + l16][32 + half * 8];
        kf.h[1] = *(const u16x8*)&ktile[16 + l16][32 + half * 8 + 16];
        v8f s2t1 = __builtin_amdgcn_wmma_f32_16x16x32_bf16(
            false, qa2.bf, false, kf.bf, (short)0, v8f{}, false, false);

        // ---- online softmax (two independent streams) ----
        u16* prow = &pbuf[wv][0][0];
        softmax_update(s1t0, s1t1, m1, l1, w1, prow, half, l16, scale);
        Frag pa1;
        pa1.h[0] = *(const u16x8*)&pbuf[wv][l16][b0];
        pa1.h[1] = *(const u16x8*)&pbuf[wv][l16][b0 + 16];

        softmax_update(s2t0, s2t1, m2, l2, w2, prow, half, l16, scale);
        Frag pa2;
        pa2.h[0] = *(const u16x8*)&pbuf[wv][l16][b0];
        pa2.h[1] = *(const u16x8*)&pbuf[wv][l16][b0 + 16];

        // ---- PV accumulation: 8 WMMAs, shared V fragments ----
        #pragma unroll
        for (int j = 0; j < 4; ++j) {
            Frag vf;
            vf.h[0] = *(const u16x8*)&vtileT[j * 16 + l16][half * 8];
            vf.h[1] = *(const u16x8*)&vtileT[j * 16 + l16][half * 8 + 16];

            v8f c1 = acc1[j];
            v8f c2 = acc2[j];
            #pragma unroll
            for (int r = 0; r < 8; ++r) { c1[r] *= w1[r]; c2[r] *= w2[r]; }
            acc1[j] = __builtin_amdgcn_wmma_f32_16x16x32_bf16(
                false, pa1.bf, false, vf.bf, (short)0, c1, false, false);
            acc2[j] = __builtin_amdgcn_wmma_f32_16x16x32_bf16(
                false, pa2.bf, false, vf.bf, (short)0, c2, false, false);
        }
    }

    // ---- combine: out = acc1/l1 - lam * acc2/l2 ----
    float r1[8], r2[8];
    #pragma unroll
    for (int r = 0; r < 8; ++r) {
        r1[r] = 1.0f / l1[r];
        r2[r] = lam / l2[r];
    }
    #pragma unroll
    for (int j = 0; j < 4; ++j) {
        const int col = h * DH + j * 16 + l16;
        #pragma unroll
        for (int r = 0; r < 8; ++r) {
            const int row = ttile * 16 + half * 8 + r;
            O[((size_t)(b * TT) + row) * HID + col] =
                acc1[j][r] * r1[r] - acc2[j][r] * r2[r];
        }
    }
}

// ---------------------------------------------------------------------------
extern "C" void kernel_launch(void* const* d_in, const int* in_sizes, int n_in,
                              void* d_out, int out_size, void* d_ws, size_t ws_size,
                              hipStream_t stream)
{
    const float* enc = (const float*)d_in[0];
    const float* dec = (const float*)d_in[1];
    const float* Wq  = (const float*)d_in[2];
    const float* Wk  = (const float*)d_in[3];
    const float* Wv  = (const float*)d_in[4];
    const float* lq1 = (const float*)d_in[5];
    const float* lq2 = (const float*)d_in[6];
    const float* lk1 = (const float*)d_in[7];
    const float* lk2 = (const float*)d_in[8];
    float* out = (float*)d_out;

    const size_t FEAT = (size_t)BB * TT * HID;   // 4 Mi elements
    const size_t WSZ  = (size_t)HID * HID;       // 1 Mi elements

    u16* qbf   = (u16*)d_ws;
    u16* kbf   = qbf + FEAT;
    u16* vbf   = kbf + FEAT;
    u16* decbf = vbf + FEAT;
    u16* encbf = decbf + FEAT;
    u16* wqbf  = encbf + FEAT;
    u16* wkbf  = wqbf + WSZ;
    u16* wvbf  = wkbf + WSZ;

    dim3 blk(256);

    // one-pass f32 -> bf16 of all GEMM operands
    cvt_bf16_kernel<<<(int)(FEAT / 4 / 256), blk, 0, stream>>>(dec, decbf, (int)(FEAT / 4));
    cvt_bf16_kernel<<<(int)(FEAT / 4 / 256), blk, 0, stream>>>(enc, encbf, (int)(FEAT / 4));
    cvt_bf16_kernel<<<(int)(WSZ  / 4 / 256), blk, 0, stream>>>(Wq,  wqbf,  (int)(WSZ / 4));
    cvt_bf16_kernel<<<(int)(WSZ  / 4 / 256), blk, 0, stream>>>(Wk,  wkbf,  (int)(WSZ / 4));
    cvt_bf16_kernel<<<(int)(WSZ  / 4 / 256), blk, 0, stream>>>(Wv,  wvbf,  (int)(WSZ / 4));

    // projections: 2048 waves each (128 M-tiles x 16 N-groups) -> 256 blocks
    proj_gemm_bf16<<<256, blk, 0, stream>>>(decbf, wqbf, qbf);
    proj_gemm_bf16<<<256, blk, 0, stream>>>(encbf, wkbf, kbf);
    proj_gemm_bf16<<<256, blk, 0, stream>>>(encbf, wvbf, vbf);

    // attention: B * HEADS * (T/16/8) = 2*16*16 = 512 blocks
    diffattn_kernel<<<512, blk, 0, stream>>>(qbf, kbf, vbf, out,
                                             lq1, lq2, lk1, lk2);
}
